// MixingAttention_27762668601561
// MI455X (gfx1250) — matrix operand order repaired and checked by reading, not compiled, not run on hardware
//
#include <hip/hip_runtime.h>
#include <hip/hip_bf16.h>
#include <hip/hip_fp16.h>

typedef __attribute__((ext_vector_type(16))) _Float16 v16h;
typedef __attribute__((ext_vector_type(8)))  _Float16 v8h;
typedef __attribute__((ext_vector_type(8)))  float    v8f;

#define WWH 12
#define WWW 12
#define TOK 144        // tokens per window (12*12)
#define NHEADS 16
#define EMB 1024       // DIM
#define CH 512         // C = DIM/2
#define HDD 32         // head dim
#define TWIN 64        // template windows
#define SWIN 256       // search windows
#define TOTB 320
#define MROWS (TOTB*TOK)   // 46080
#define TROWS (TWIN*TOK)   // 9216

// ---------------- WMMA helpers (CDNA5 16x16x32 f16 -> f32) ----------------

__device__ __forceinline__ v8f wmma16(v16h a, v16h b, v8f c) {
  return __builtin_amdgcn_wmma_f32_16x16x32_f16(false, a, false, b, (short)0, c,
                                                false, false);
}

// A fragment, row-major 16xK f16 tile at A, leading dim lda (halves).
// lane<16: M=lane, elems = K[0..7],K[16..23]; lane>=16: M=lane-16, K[8..15],K[24..31]
__device__ __forceinline__ v16h frag_a(const _Float16* A, int lda) {
  const int lane = threadIdx.x & 31;
  const int row  = lane & 15;
  const int k0   = (lane < 16) ? 0 : 8;
  union { v16h v; v8h h[2]; } u;
  u.h[0] = *(const v8h*)(A + (long)row * lda + k0);
  u.h[1] = *(const v8h*)(A + (long)row * lda + k0 + 16);
  return u.v;
}

// A fragment loaded directly from a row-major f32 tile with in-register f16
// conversion (no LDS staging, no barriers). Two contiguous 8-float runs/lane.
__device__ __forceinline__ v16h frag_a_cvt(const float* A, int lda) {
  const int lane = threadIdx.x & 31;
  const int row  = lane & 15;
  const int k0   = (lane < 16) ? 0 : 8;
  const float* p = A + (long)row * lda + k0;
  v16h r;
#pragma unroll
  for (int i = 0; i < 8; ++i) r[i] = (_Float16)p[i];
#pragma unroll
  for (int i = 0; i < 8; ++i) r[8 + i] = (_Float16)p[16 + i];
  return r;
}

// B fragment from a K-contiguous ("transposed", [N][K]) buffer; BT points at
// column 0 of the 16-col tile, ldk = leading dim in halves.
// lane<16: N=lane, K[0..7],K[16..23]; lane>=16: N=lane-16, K[8..15],K[24..31]
__device__ __forceinline__ v16h frag_bT(const _Float16* BT, int ldk) {
  const int lane = threadIdx.x & 31;
  const int col  = lane & 15;
  const int k0   = (lane < 16) ? 0 : 8;
  union { v16h v; v8h h[2]; } u;
  u.h[0] = *(const v8h*)(BT + (long)col * ldk + k0);
  u.h[1] = *(const v8h*)(BT + (long)col * ldk + k0 + 16);
  return u.v;
}

// ---------------- prep kernels ----------------

// W[K][N] f32 -> WT[N][K] f16 (K-contiguous for B fragments)
__global__ void k_w_transpose(const float* __restrict__ W,
                              _Float16* __restrict__ WT, int K, int N) {
  long t = (long)blockIdx.x * blockDim.x + threadIdx.x;
  if (t >= (long)K * N) return;
  int k = (int)(t / N), n = (int)(t % N);
  WT[(long)n * K + k] = (_Float16)W[(long)k * N + n];
}

// expand relative-position bias table to [h][n][m] f32
__global__ void k_bias_expand(const float* __restrict__ rpb,
                              float* __restrict__ biasbuf) {
  int t = blockIdx.x * blockDim.x + threadIdx.x;
  if (t >= NHEADS * TOK * TOK) return;
  int m = t % TOK, n = (t / TOK) % TOK, h = t / (TOK * TOK);
  int yn = n / WWW, xn = n % WWW, ym = m / WWW, xm = m % WWW;
  int idx = (yn - ym + WWH - 1) * (2 * WWW - 1) + (xn - xm + WWW - 1);
  biasbuf[t] = rpb[idx * NHEADS + h];
}

// ---------------- stage 1: projection + layernorm -> xa (f16) ----------------
// block = 16 rows x 512 cols, 8 waves x 64 cols; K=1024.
// A fragments come straight from global f32 with in-register conversion, so
// the K loop is barrier-free; the only block sync is the LN reduction.
__launch_bounds__(256)
__global__ void k_proj_ln(const float* __restrict__ x, const float* __restrict__ tpl,
                          const _Float16* __restrict__ wpsT, const float* __restrict__ bps,
                          const float* __restrict__ gps,  const float* __restrict__ beps,
                          const _Float16* __restrict__ wptT, const float* __restrict__ bpt,
                          const float* __restrict__ gpt,  const float* __restrict__ bept,
                          _Float16* __restrict__ xa) {
  __shared__ float red[8][16][2];
  __shared__ float mv[16][2];
  const int tid = threadIdx.x, w = tid >> 5, lane = tid & 31;
  const int r0 = blockIdx.x * 16;
  const bool is_t = (r0 < TROWS);
  const float*     src  = is_t ? (tpl + (long)r0 * EMB) : (x + (long)(r0 - TROWS) * EMB);
  const _Float16*  WT   = is_t ? wptT : wpsT;
  const float*     bias = is_t ? bpt  : bps;
  const float*     gg   = is_t ? gpt  : gps;
  const float*     bb   = is_t ? bept : beps;
  const int cb = w * 64;
  v8f acc[4] = {};
  for (int k0 = 0; k0 < EMB; k0 += 32) {
    v16h a = frag_a_cvt(src + k0, EMB);
#pragma unroll
    for (int j = 0; j < 4; ++j) {
      v16h b = frag_bT(WT + (long)(cb + j * 16) * EMB + k0, EMB);
      acc[j] = wmma16(a, b, acc[j]);
    }
  }
  // + bias, LN stats
  float gv[4], bev[4];
#pragma unroll
  for (int j = 0; j < 4; ++j) {
    int col = cb + j * 16 + (lane & 15);
    float bj = bias[col];
    gv[j] = gg[col]; bev[j] = bb[col];
#pragma unroll
    for (int i = 0; i < 8; ++i) acc[j][i] += bj;
  }
#pragma unroll
  for (int i = 0; i < 8; ++i) {
    float s = acc[0][i] + acc[1][i] + acc[2][i] + acc[3][i];
    float q = acc[0][i] * acc[0][i] + acc[1][i] * acc[1][i] +
              acc[2][i] * acc[2][i] + acc[3][i] * acc[3][i];
#pragma unroll
    for (int m = 1; m < 16; m <<= 1) {     // 16-lane halves = distinct rows
      s += __shfl_xor(s, m, 32);
      q += __shfl_xor(q, m, 32);
    }
    if (lane == 0)  { red[w][i][0] = s;     red[w][i][1] = q; }
    if (lane == 16) { red[w][i + 8][0] = s; red[w][i + 8][1] = q; }
  }
  __syncthreads();
  if (tid < 16) {
    float s = 0.f, q = 0.f;
    for (int ww = 0; ww < 8; ++ww) { s += red[ww][tid][0]; q += red[ww][tid][1]; }
    float mu  = s / (float)CH;
    float var = q / (float)CH - mu * mu;
    mv[tid][0] = mu;
    mv[tid][1] = rsqrtf(var + 1e-5f);
  }
  __syncthreads();
#pragma unroll
  for (int i = 0; i < 8; ++i) {
    int rl = (lane < 16) ? i : i + 8;
    float mu = mv[rl][0], rs = mv[rl][1];
    long row = r0 + rl;
#pragma unroll
    for (int j = 0; j < 4; ++j) {
      int col = cb + j * 16 + (lane & 15);
      float y = (acc[j][i] - mu) * rs * gv[j] + bev[j];
      xa[row * CH + col] = (_Float16)y;
    }
  }
}

// ---------------- stage 2: QKV GEMM, scatter into [b][h][n][d] f16 ----------------
__launch_bounds__(128)
__global__ void k_qkv(const _Float16* __restrict__ xa, const _Float16* __restrict__ wT,
                      const float* __restrict__ bqkv,
                      _Float16* __restrict__ qb, _Float16* __restrict__ kb,
                      _Float16* __restrict__ vb) {
  const int tid = threadIdx.x, w = tid >> 5, lane = tid & 31;
  const int r0 = blockIdx.x * 16;
  const int cb = blockIdx.y * 256 + w * 64;
  const float scale = 0.1767766952966369f;  // HEAD_DIM^-0.5
  v8f acc[4] = {};
  const _Float16* A = xa + (long)r0 * CH;
  for (int k0 = 0; k0 < CH; k0 += 32) {
    v16h a = frag_a(A + k0, CH);
#pragma unroll
    for (int j = 0; j < 4; ++j) {
      v16h b = frag_bT(wT + (long)(cb + j * 16) * CH + k0, CH);
      acc[j] = wmma16(a, b, acc[j]);
    }
  }
#pragma unroll
  for (int j = 0; j < 4; ++j) {
    int c = cb + j * 16 + (lane & 15);
    int which = c >> 9, rem = c & 511, h = rem >> 5, d = rem & 31;
    float bv = bqkv[c];
    _Float16* dst = (which == 0) ? qb : (which == 1) ? kb : vb;
    float sc = (which == 0) ? scale : 1.0f;
#pragma unroll
    for (int i = 0; i < 8; ++i) {
      int rl = (lane < 16) ? i : i + 8;
      int row = r0 + rl;
      int bidx = row / TOK, n = row % TOK;
      dst[(((long)bidx * NHEADS + h) * TOK + n) * HDD + d] =
          (_Float16)((acc[j][i] + bv) * sc);
    }
  }
}

// ---------------- stage 3: attention, one block per (b,h) ----------------
__launch_bounds__(128)
__global__ void k_attn(const _Float16* __restrict__ qb, const _Float16* __restrict__ kb,
                       const _Float16* __restrict__ vb, const float* __restrict__ biasbuf,
                       const float* __restrict__ mask0, const float* __restrict__ mask1,
                       _Float16* __restrict__ aout) {
  __shared__ _Float16 qs[TOK][HDD];     //  9 KB
  __shared__ _Float16 ks[TOK][HDD];     //  9 KB
  __shared__ _Float16 vt[HDD][160];     // 10 KB, V transposed, K padded 144->160
  __shared__ _Float16 ps[4][16][160];   // 20 KB, per-wave P tiles
  const int tid = threadIdx.x, w = tid >> 5, lane = tid & 31;
  const int bh = blockIdx.x, b = bh >> 4, h = bh & 15;
  const long base = (long)bh * TOK * HDD;
  for (int i = tid; i < TOK * HDD / 2; i += 128) {
    ((unsigned int*)&qs[0][0])[i] = ((const unsigned int*)(qb + base))[i];
    ((unsigned int*)&ks[0][0])[i] = ((const unsigned int*)(kb + base))[i];
  }
  for (int i = tid; i < TOK * HDD; i += 128) {
    int n = i >> 5, d = i & 31;
    vt[d][n] = vb[base + i];
  }
  for (int i = tid; i < HDD * 16; i += 128)            // zero V K-pad
    vt[i >> 4][TOK + (i & 15)] = (_Float16)0.f;
  for (int i = tid; i < 4 * 16 * 16; i += 128) {       // zero P K-pad
    int wq = i >> 8, r = (i >> 4) & 15, c = i & 15;
    ps[wq][r][TOK + c] = (_Float16)0.f;
  }
  __syncthreads();

  const float* biasH = biasbuf + (long)h * TOK * TOK;
  const float* maskp = (b < TWIN) ? (mask1 + (long)(b & 3) * TOK * TOK)
                                  : (mask0 + (long)((b - TWIN) & 15) * TOK * TOK);
  for (int rt = w; rt < 9; rt += 4) {
    // S = q @ k^T : 9 tiles of 16x16, single K=32 WMMA each
    v16h aq = frag_a(&qs[rt * 16][0], HDD);
    v8f s[9];
#pragma unroll
    for (int ct = 0; ct < 9; ++ct) {
      v16h bk = frag_bT(&ks[ct * 16][0], HDD);
      v8f z = {};
      s[ct] = wmma16(aq, bk, z);
    }
    // + rel-pos bias + window mask
#pragma unroll
    for (int ct = 0; ct < 9; ++ct) {
      int m = ct * 16 + (lane & 15);
#pragma unroll
      for (int i = 0; i < 8; ++i) {
        int n = rt * 16 + ((lane < 16) ? i : i + 8);
        s[ct][i] += biasH[n * TOK + m] + maskp[n * TOK + m];
      }
    }
    // row softmax (rows split across the two 16-lane halves)
#pragma unroll
    for (int i = 0; i < 8; ++i) {
      float mx = s[0][i];
#pragma unroll
      for (int ct = 1; ct < 9; ++ct) mx = fmaxf(mx, s[ct][i]);
#pragma unroll
      for (int mm = 1; mm < 16; mm <<= 1) mx = fmaxf(mx, __shfl_xor(mx, mm, 32));
      float sum = 0.f;
#pragma unroll
      for (int ct = 0; ct < 9; ++ct) { s[ct][i] = __expf(s[ct][i] - mx); sum += s[ct][i]; }
#pragma unroll
      for (int mm = 1; mm < 16; mm <<= 1) sum += __shfl_xor(sum, mm, 32);
      float inv = 1.0f / sum;
      int rl = (lane < 16) ? i : i + 8;
#pragma unroll
      for (int ct = 0; ct < 9; ++ct)
        ps[w][rl][ct * 16 + (lane & 15)] = (_Float16)(s[ct][i] * inv);
    }
    // out = P @ V : K = 160 (padded), 5 WMMAs per 16-col tile
#pragma unroll
    for (int nt = 0; nt < 2; ++nt) {
      v8f o = {};
#pragma unroll
      for (int kt = 0; kt < 5; ++kt) {
        v16h ap = frag_a(&ps[w][0][kt * 32], 160);
        v16h bv = frag_bT(&vt[nt * 16][0] + kt * 32, 160);
        o = wmma16(ap, bv, o);
      }
#pragma unroll
      for (int i = 0; i < 8; ++i) {
        int rl = (lane < 16) ? i : i + 8;
        int n = rt * 16 + rl;
        int col = h * HDD + nt * 16 + (lane & 15);
        aout[((long)b * TOK + n) * CH + col] = (_Float16)o[i];
      }
    }
  }
}

// ---------------- stage 4: output projection (f32 out) ----------------
__launch_bounds__(128)
__global__ void k_oproj(const _Float16* __restrict__ aout,
                        const _Float16* __restrict__ wtrT, const float* __restrict__ btr,
                        const _Float16* __restrict__ wsrT, const float* __restrict__ bsr,
                        float* __restrict__ out) {
  const int tid = threadIdx.x, w = tid >> 5, lane = tid & 31;
  const int r0 = blockIdx.x * 16;
  const int cb = blockIdx.y * 256 + w * 64;
  const bool is_t = r0 < TROWS;
  const _Float16* WT = is_t ? wtrT : wsrT;
  const float* bias  = is_t ? btr  : bsr;
  v8f acc[4] = {};
  const _Float16* A = aout + (long)r0 * CH;
  for (int k0 = 0; k0 < CH; k0 += 32) {
    v16h a = frag_a(A + k0, CH);
#pragma unroll
    for (int j = 0; j < 4; ++j) {
      v16h b = frag_bT(WT + (long)(cb + j * 16) * CH + k0, CH);
      acc[j] = wmma16(a, b, acc[j]);
    }
  }
#pragma unroll
  for (int j = 0; j < 4; ++j) {
    int c = cb + j * 16 + (lane & 15);
    float bv = bias[c];
#pragma unroll
    for (int i = 0; i < 8; ++i) {
      int rl = (lane < 16) ? i : i + 8;
      long row = r0 + rl;
      out[row * EMB + c] = acc[j][i] + bv;  // template rows then search rows = flat output
    }
  }
}

// ---------------- host ----------------

extern "C" void kernel_launch(void* const* d_in, const int* in_sizes, int n_in,
                              void* d_out, int out_size, void* d_ws, size_t ws_size,
                              hipStream_t stream) {
  const float* x     = (const float*)d_in[0];
  const float* tpl   = (const float*)d_in[1];
  const float* mask0 = (const float*)d_in[2];
  const float* mask1 = (const float*)d_in[3];
  const float* rpb   = (const float*)d_in[8];
  const float* w_ps  = (const float*)d_in[9];
  const float* b_ps  = (const float*)d_in[10];
  const float* g_ps  = (const float*)d_in[11];
  const float* be_ps = (const float*)d_in[12];
  const float* w_pt  = (const float*)d_in[13];
  const float* b_pt  = (const float*)d_in[14];
  const float* g_pt  = (const float*)d_in[15];
  const float* be_pt = (const float*)d_in[16];
  const float* w_qkv = (const float*)d_in[17];
  const float* b_qkv = (const float*)d_in[18];
  const float* w_tr  = (const float*)d_in[19];
  const float* b_tr  = (const float*)d_in[20];
  const float* w_sr  = (const float*)d_in[21];
  const float* b_sr  = (const float*)d_in[22];
  float* out = (float*)d_out;

  // workspace carve (256B aligned)
  char* cur = (char*)d_ws;
  auto take = [&](size_t bytes) {
    void* p = cur;
    cur += (bytes + 255) & ~(size_t)255;
    return p;
  };
  _Float16* wpsT  = (_Float16*)take((size_t)CH * EMB * 2);
  _Float16* wptT  = (_Float16*)take((size_t)CH * EMB * 2);
  _Float16* wqkvT = (_Float16*)take((size_t)(3 * CH) * CH * 2);
  _Float16* wtrT  = (_Float16*)take((size_t)EMB * CH * 2);
  _Float16* wsrT  = (_Float16*)take((size_t)EMB * CH * 2);
  float*    biasbuf = (float*)take((size_t)NHEADS * TOK * TOK * 4);
  _Float16* xa = (_Float16*)take((size_t)MROWS * CH * 2);
  _Float16* qb = (_Float16*)take((size_t)MROWS * CH * 2);
  _Float16* kb = (_Float16*)take((size_t)MROWS * CH * 2);
  _Float16* vb = (_Float16*)take((size_t)MROWS * CH * 2);
  _Float16* aout = xa;  // xa is dead after k_qkv; reuse for attention output

  // weight transposes + f16 casts
  {
    int thr = 256;
    k_w_transpose<<<(EMB * CH + thr - 1) / thr, thr, 0, stream>>>(w_ps, wpsT, EMB, CH);
    k_w_transpose<<<(EMB * CH + thr - 1) / thr, thr, 0, stream>>>(w_pt, wptT, EMB, CH);
    k_w_transpose<<<(CH * 3 * CH + thr - 1) / thr, thr, 0, stream>>>(w_qkv, wqkvT, CH, 3 * CH);
    k_w_transpose<<<(CH * EMB + thr - 1) / thr, thr, 0, stream>>>(w_tr, wtrT, CH, EMB);
    k_w_transpose<<<(CH * EMB + thr - 1) / thr, thr, 0, stream>>>(w_sr, wsrT, CH, EMB);
    k_bias_expand<<<(NHEADS * TOK * TOK + thr - 1) / thr, thr, 0, stream>>>(rpb, biasbuf);
  }

  // stage 1: proj + LN -> xa (f16)
  k_proj_ln<<<MROWS / 16, 256, 0, stream>>>(x, tpl, wpsT, b_ps, g_ps, be_ps,
                                            wptT, b_pt, g_pt, be_pt, xa);
  // stage 2: qkv
  k_qkv<<<dim3(MROWS / 16, (3 * CH) / 256), 128, 0, stream>>>(xa, wqkvT, b_qkv,
                                                              qb, kb, vb);
  // stage 3: attention per (b,h)
  k_attn<<<TOTB * NHEADS, 128, 0, stream>>>(qb, kb, vb, biasbuf, mask0, mask1, aout);
  // stage 4: output projections (writes template rows then search rows = d_out layout)
  k_oproj<<<dim3(MROWS / 16, EMB / 256), 128, 0, stream>>>(aout, wtrT, b_tr,
                                                           wsrT, b_sr, out);
}
